// EgoGCN_4252017623441
// MI455X (gfx1250) — matrix operand order repaired and checked by reading, not compiled, not run on hardware
//
#include <hip/hip_runtime.h>
#include <hip/hip_bf16.h>

typedef __attribute__((ext_vector_type(16))) __bf16     v16bf;
typedef __attribute__((ext_vector_type(8)))  float      v8f;
typedef __attribute__((ext_vector_type(4)))  unsigned   v4u;

union BfFrag {
    v16bf v;
    v4u   q[2];
};

static __device__ __forceinline__ unsigned short f32_to_bf16(float f) {
    unsigned u = __float_as_uint(f);
    unsigned r = (u + 0x7FFFu + ((u >> 16) & 1u)) >> 16;   // round-to-nearest-even
    return (unsigned short)r;
}

// ---------------------------------------------------------------------------
// Prep: W1 (K x N, K<=32, zero-padded to 32) -> W1T bf16 [N][32]
// ---------------------------------------------------------------------------
__global__ void pack_w1t(const float* __restrict__ W1, unsigned short* __restrict__ W1T,
                         int K, int N) {
    int t = blockIdx.x * blockDim.x + threadIdx.x;
    if (t >= N * 32) return;
    int n = t >> 5, k = t & 31;
    float v = (k < K) ? W1[k * N + n] : 0.0f;
    W1T[n * 32 + k] = f32_to_bf16(v);
}

// W2 (1024 x 1024) -> W2T bf16 [n][k]
__global__ void pack_w2t(const float* __restrict__ W2, unsigned short* __restrict__ W2T) {
    int t = blockIdx.x * blockDim.x + threadIdx.x;
    if (t >= 1024 * 1024) return;
    int n = t >> 10, k = t & 1023;
    W2T[n * 1024 + k] = f32_to_bf16(W2[k * 1024 + n]);
}

// ---------------------------------------------------------------------------
// Fused NNConv edge kernel. One wave (32 lanes) owns 16 edges:
//   H1 = relu(EA @ W1 + b1)            (16 x 1024)  -- 64 WMMA
//   H2 = H1 @ W2 + b2                  (16 x 1024)  -- 2048 WMMA, register-blocked
//   msg[e][o] = sum_i x[src[e]][i] * H2[e][i*32+o]  -- fused from C fragments
//   atomic scatter msg into seg[dst], count into cnt[dst]
// ---------------------------------------------------------------------------
__global__ __launch_bounds__(32) void nnconv_edge_kernel(
    const float* __restrict__ x, const float* __restrict__ ea,
    const int* __restrict__ src, const int* __restrict__ dst,
    const unsigned short* __restrict__ W1T, const float* __restrict__ b1,
    const unsigned short* __restrict__ W2T, const float* __restrict__ b2,
    float* __restrict__ seg, float* __restrict__ cnt, int n_edges)
{
    __shared__ __align__(16) unsigned short ldsA[16 * 32];     // EA tile, bf16, K padded
    __shared__ __align__(16) unsigned short ldsH1[16 * 1024];  // H1 tile, bf16
    __shared__ __align__(16) float          ldsX[16 * 32];     // x[src] tile, f32
    __shared__ int ldsDst[16];
    __shared__ int ldsSrc[16];

    const int lane = threadIdx.x;
    const int e0   = blockIdx.x * 16;
    const int ln   = lane & 15;
    const int hi   = lane >> 4;          // 0 | 1 (wave half)
    const int m0   = ln;                 // A-fragment row for this lane
    const int kq   = hi ? 8 : 0;         // unused; A fragments use kb as well
    const int kb   = hi ? 16 : 0;        // B-fragment K base
    const int mh   = hi ? 8 : 0;         // C-fragment row base

    // --- stage edge_attr (zero-padded K=32), indices, x gather tile ---------
    for (int i = lane; i < 16 * 32; i += 32) ldsA[i] = 0;
    __syncthreads();
    for (int i = lane; i < 16 * 16; i += 32) {
        int r = i >> 4, c = i & 15;
        int e = e0 + r;
        float v = (e < n_edges) ? ea[e * 16 + c] : 0.0f;
        ldsA[r * 32 + c] = f32_to_bf16(v);
    }
    if (lane < 16) {
        int e = e0 + lane;
        int sv = 0, dv = 0;
        if (e < n_edges) {
            sv = src[e]; dv = dst[e];
            atomicAdd(&cnt[dv], 1.0f);
        }
        ldsSrc[lane] = sv;
        ldsDst[lane] = dv;
    }
    __syncthreads();
    for (int r = 0; r < 16; ++r)
        ldsX[r * 32 + lane] = x[ldsSrc[r] * 32 + lane];
    __syncthreads();

    // --- GEMM1: H1 = relu(EA @ W1 + b1) -------------------------------------
    BfFrag a1;
    a1.q[0] = *(const v4u*)&ldsA[m0 * 32 + kb];
    a1.q[1] = *(const v4u*)&ldsA[m0 * 32 + kb + 8];

    for (int j = 0; j < 64; ++j) {
        const int col = j * 16 + ln;
        BfFrag b;
        const v4u* wp = (const v4u*)(W1T + col * 32 + kb);
        b.q[0] = wp[0]; b.q[1] = wp[1];
        v8f c = {};
        c = __builtin_amdgcn_wmma_f32_16x16x32_bf16(false, a1.v, false, b.v,
                                                    (short)0, c, false, false);
        const float bb = b1[col];
#pragma unroll
        for (int r = 0; r < 8; ++r) {
            float v = c[r] + bb;
            v = v > 0.0f ? v : 0.0f;
            ldsH1[(r + mh) * 1024 + col] = f32_to_bf16(v);
        }
    }
    __syncthreads();

    // --- GEMM2 register-blocked (8 n-tiles / pass) + fused message contract --
    float accLo[8], accHi[8];
#pragma unroll
    for (int r = 0; r < 8; ++r) { accLo[r] = 0.0f; accHi[r] = 0.0f; }

    for (int jg = 0; jg < 8; ++jg) {
        v8f c[8];
#pragma unroll
        for (int jj = 0; jj < 8; ++jj) c[jj] = {};

        const unsigned short* wbase = W2T + ((jg * 8) * 16 + ln) * 1024 + kb;

        for (int kk = 0; kk < 32; ++kk) {
            BfFrag a;
            a.q[0] = *(const v4u*)&ldsH1[m0 * 1024 + kk * 32 + kb];
            a.q[1] = *(const v4u*)&ldsH1[m0 * 1024 + kk * 32 + kb + 8];
#pragma unroll
            for (int jj = 0; jj < 8; ++jj) {
                BfFrag b;
                const v4u* wp = (const v4u*)(wbase + (size_t)jj * 16 * 1024 + kk * 32);
                b.q[0] = wp[0]; b.q[1] = wp[1];
                c[jj] = __builtin_amdgcn_wmma_f32_16x16x32_bf16(
                    false, a.v, false, b.v, (short)0, c[jj], false, false);
            }
        }

        // consume the 8 finished 16x16 tiles into message accumulators
#pragma unroll
        for (int jj = 0; jj < 8; ++jj) {
            const int j   = jg * 8 + jj;
            const int col = j * 16 + ln;
            const float bb = b2[col];
            const int   i  = j >> 1;                  // input-feature index
            float* acc = (j & 1) ? accHi : accLo;     // o = ln | ln+16
#pragma unroll
            for (int r = 0; r < 8; ++r) {
                float h2 = c[jj][r] + bb;
                acc[r] += ldsX[(r + mh) * 32 + i] * h2;
            }
        }
    }

    // --- scatter into segment sums ------------------------------------------
#pragma unroll
    for (int r = 0; r < 8; ++r) {
        int m = r + mh;
        int e = e0 + m;
        if (e < n_edges) {
            int d = ldsDst[m];
            atomicAdd(&seg[d * 32 + ln],      accLo[r]);
            atomicAdd(&seg[d * 32 + 16 + ln], accHi[r]);
        }
    }
}

// ---------------------------------------------------------------------------
// Combine: out = relu(seg/max(cnt,1) + x @ Wroot + bias)
// ---------------------------------------------------------------------------
__global__ void combine_kernel(const float* __restrict__ xin,
                               const float* __restrict__ Wroot,
                               const float* __restrict__ bias,
                               const float* __restrict__ seg,
                               const float* __restrict__ cnt,
                               float* __restrict__ out, int n_nodes)
{
    __shared__ float Wr[32 * 32];
    for (int i = threadIdx.x; i < 1024; i += blockDim.x) Wr[i] = Wroot[i];
    __syncthreads();
    int t = blockIdx.x * blockDim.x + threadIdx.x;
    if (t >= n_nodes * 32) return;
    int v = t >> 5, c = t & 31;
    float s = bias[c];
    const float* xr = xin + v * 32;
#pragma unroll 8
    for (int i = 0; i < 32; ++i) s += xr[i] * Wr[i * 32 + c];
    float cn = cnt[v];
    cn = cn > 1.0f ? cn : 1.0f;
    s += seg[t] / cn;
    out[t] = s > 0.0f ? s : 0.0f;
}

// ---------------------------------------------------------------------------
// Classifier: out[g] = h[ptr[g]] @ Wcls + bcls
// ---------------------------------------------------------------------------
__global__ void cls_kernel(const float* __restrict__ h, const int* __restrict__ ptr,
                           const float* __restrict__ Wcls, const float* __restrict__ bcls,
                           float* __restrict__ out, int n_graphs)
{
    int t = blockIdx.x * blockDim.x + threadIdx.x;
    if (t >= n_graphs * 16) return;
    int g = t >> 4, c = t & 15;
    int v = ptr[g];
    float s = bcls[c];
    const float* hr = h + v * 32;
#pragma unroll 8
    for (int i = 0; i < 32; ++i) s += hr[i] * Wcls[i * 16 + c];
    out[t] = s;
}

// ---------------------------------------------------------------------------
extern "C" void kernel_launch(void* const* d_in, const int* in_sizes, int n_in,
                              void* d_out, int out_size, void* d_ws, size_t ws_size,
                              hipStream_t stream)
{
    const float* x     = (const float*)d_in[0];
    const float* ea    = (const float*)d_in[1];
    const int*   src   = (const int*)d_in[2];
    const int*   dst   = (const int*)d_in[3];
    const int*   ptr   = (const int*)d_in[4];
    const float* W1a   = (const float*)d_in[5];
    const float* b1a   = (const float*)d_in[6];
    const float* W2a   = (const float*)d_in[7];
    const float* b2a   = (const float*)d_in[8];
    const float* roota = (const float*)d_in[9];
    const float* biasa = (const float*)d_in[10];
    const float* W1b   = (const float*)d_in[11];
    const float* b1b   = (const float*)d_in[12];
    const float* W2b   = (const float*)d_in[13];
    const float* b2b   = (const float*)d_in[14];
    const float* rootb = (const float*)d_in[15];
    const float* biasb = (const float*)d_in[16];
    const float* Wcls  = (const float*)d_in[17];
    const float* bcls  = (const float*)d_in[18];
    float* out = (float*)d_out;

    const int n_nodes  = in_sizes[0] / 32;
    const int n_edges  = in_sizes[2];
    const int n_graphs = in_sizes[4] - 1;

    // workspace carve-up (256B aligned)
    char* ws = (char*)d_ws;
    size_t off = 0;
    auto carve = [&](size_t bytes) {
        char* p = ws + off;
        off = (off + bytes + 255) & ~(size_t)255;
        return p;
    };
    unsigned short* W1Ta = (unsigned short*)carve(1024 * 32 * 2);
    unsigned short* W1Tb = (unsigned short*)carve(1024 * 32 * 2);
    unsigned short* W2Ta = (unsigned short*)carve(1024 * 1024 * 2);
    unsigned short* W2Tb = (unsigned short*)carve(1024 * 1024 * 2);
    float* seg = (float*)carve((size_t)n_nodes * 32 * 4);
    float* cnt = (float*)carve((size_t)n_nodes * 4);
    float* h1  = (float*)carve((size_t)n_nodes * 32 * 4);
    float* h2  = (float*)carve((size_t)n_nodes * 32 * 4);

    // pack weights to transposed bf16 (fits in L2: 2 MB each)
    pack_w1t<<<(1024 * 32 + 255) / 256, 256, 0, stream>>>(W1a, W1Ta, 16, 1024);
    pack_w1t<<<(1024 * 32 + 255) / 256, 256, 0, stream>>>(W1b, W1Tb, 16, 1024);
    pack_w2t<<<(1024 * 1024 + 255) / 256, 256, 0, stream>>>(W2a, W2Ta);
    pack_w2t<<<(1024 * 1024 + 255) / 256, 256, 0, stream>>>(W2b, W2Tb);

    const int edge_blocks = (n_edges + 15) / 16;
    const int comb_blocks = (n_nodes * 32 + 255) / 256;

    // ---- layer 1 ----
    hipMemsetAsync(seg, 0, (size_t)n_nodes * 32 * 4, stream);
    hipMemsetAsync(cnt, 0, (size_t)n_nodes * 4, stream);
    nnconv_edge_kernel<<<edge_blocks, 32, 0, stream>>>(
        x, ea, src, dst, W1Ta, b1a, W2Ta, b2a, seg, cnt, n_edges);
    combine_kernel<<<comb_blocks, 256, 0, stream>>>(
        x, roota, biasa, seg, cnt, h1, n_nodes);

    // ---- layer 2 ----
    hipMemsetAsync(seg, 0, (size_t)n_nodes * 32 * 4, stream);
    hipMemsetAsync(cnt, 0, (size_t)n_nodes * 4, stream);
    nnconv_edge_kernel<<<edge_blocks, 32, 0, stream>>>(
        h1, ea, src, dst, W1Tb, b1b, W2Tb, b2b, seg, cnt, n_edges);
    combine_kernel<<<comb_blocks, 256, 0, stream>>>(
        h1, rootb, biasb, seg, cnt, h2, n_nodes);

    // ---- classifier ----
    cls_kernel<<<(n_graphs * 16 + 255) / 256, 256, 0, stream>>>(
        h2, ptr, Wcls, bcls, out, n_graphs);
}
